// GQA_Einsum_22308060136005
// MI455X (gfx1250) — matrix operand order repaired and checked by reading, not compiled, not run on hardware
//
#include <hip/hip_runtime.h>
#include <math.h>

// ---------------------------------------------------------------------------
// GQA attention for MI455X (gfx1250, wave32, WMMA)
//   B=2, L=2048, E=2048, Hq=32, Hkv=8, D=64, group=4
// Pipeline: f32->bf16 convert, QKV GEMMs (v_wmma_f32_16x16x32_bf16),
// fused flash-style attention (async global->LDS staging + WMMA + online
// softmax), O GEMM -> f32.
// ---------------------------------------------------------------------------

typedef __bf16          v16bf __attribute__((ext_vector_type(16)));
typedef float           v8f   __attribute__((ext_vector_type(8)));
typedef float           f4    __attribute__((ext_vector_type(4)));
typedef unsigned short  us4   __attribute__((ext_vector_type(4)));
typedef unsigned short  us8   __attribute__((ext_vector_type(8)));
typedef int             b128v __attribute__((vector_size(16)));   // builtin's B128 type

union ABFrag {            // one WMMA 16-bit A/B operand: 16 bf16 per lane
  us8   u[2];
  v16bf v;
};

__device__ __forceinline__ unsigned short f2bf(float f) {
  unsigned int x = __builtin_bit_cast(unsigned int, f);
  x += 0x7FFFu + ((x >> 16) & 1u);          // round-to-nearest-even
  return (unsigned short)(x >> 16);
}

__device__ __forceinline__ v8f wmma_bf16(v16bf a, v16bf b, v8f c) {
  // (neg_a, A, neg_b, B, c_mod, C, reuse_a, reuse_b)
  return __builtin_amdgcn_wmma_f32_16x16x32_bf16(false, a, false, b,
                                                 (short)0, c, false, false);
}

// --- CDNA5 async global->LDS path (ASYNCcnt), guarded for portability ------
#if defined(__gfx1250__) &&                                        \
    __has_builtin(__builtin_amdgcn_global_load_async_to_lds_b128) && \
    __has_builtin(__builtin_amdgcn_s_wait_asynccnt)
#define USE_ASYNC_LDS 1
#endif

__device__ __forceinline__ void copy16_g2lds(const unsigned short* g,
                                             unsigned short* l) {
#ifdef USE_ASYNC_LDS
  // global_load_async_to_lds_b128(global b128*, lds b128*, imm offset, imm cpol)
  __builtin_amdgcn_global_load_async_to_lds_b128(
      (__attribute__((address_space(1))) b128v*)g,
      (__attribute__((address_space(3))) b128v*)l, 0, 0);
#else
  *(us8*)l = *(const us8*)g;
#endif
}

__device__ __forceinline__ void wait_async_copies() {
#ifdef USE_ASYNC_LDS
  __builtin_amdgcn_s_wait_asynccnt(0);
#endif
}

// ---------------------------------------------------------------------------
// f32 -> bf16 conversion, 4 elements/thread
// ---------------------------------------------------------------------------
__global__ __launch_bounds__(256)
void cvt_f32_to_bf16(const float* __restrict__ s,
                     unsigned short* __restrict__ d, int n4) {
  int i = blockIdx.x * 256 + threadIdx.x;
  if (i >= n4) return;
  f4 v = ((const f4*)s)[i];
  us4 o = { f2bf(v[0]), f2bf(v[1]), f2bf(v[2]), f2bf(v[3]) };
  ((us4*)d)[i] = o;
}

// ---------------------------------------------------------------------------
// C[M,N] = scale * (A[M,K] @ W[N,K]^T), bf16 inputs, f32 accumulation.
// 256 threads = 8 waves; block tile 128x128; wave tile 64x32 (4x2 WMMA accums).
// Operands fetched straight from global: whole bf16 working set (~30 MB)
// resides in the 192 MB L2, so re-reads across block columns hit L2.
// ---------------------------------------------------------------------------
template <int OUT_F32>
__global__ __launch_bounds__(256)
void gemm_bf16_wmma(const unsigned short* __restrict__ A,
                    const unsigned short* __restrict__ W,
                    void* __restrict__ C,
                    int M, int N, int K, float scale) {
  const int lane = threadIdx.x & 31;
  const int wv   = threadIdx.x >> 5;
  const int hlf  = lane >> 4;     // lane half-group (A/B K-offset select)
  const int l16  = lane & 15;
  const int bm   = blockIdx.y * 128 + (wv >> 2) * 64;
  const int bn   = blockIdx.x * 128 + (wv & 3) * 32;

  const v8f vzero = {0.f,0.f,0.f,0.f,0.f,0.f,0.f,0.f};
  v8f acc[4][2];
#pragma unroll
  for (int i = 0; i < 4; ++i)
#pragma unroll
    for (int j = 0; j < 2; ++j) acc[i][j] = vzero;

  // A 16-bit operand map: lane half 0 -> K [0..7]+[16..23], half 1 -> +8
  const unsigned short* Abase = A + (size_t)(bm + l16) * K + hlf * 8;
  // B 16-bit operand map: lane half 0 -> K [0..7]+[16..23], half 1 -> +8
  const unsigned short* Wbase = W + (size_t)(bn + l16) * K + hlf * 8;

  for (int k0 = 0; k0 < K; k0 += 32) {
    ABFrag a[4], b[2];
#pragma unroll
    for (int mt = 0; mt < 4; ++mt) {
      const unsigned short* p = Abase + (size_t)mt * 16 * K + k0;
      a[mt].u[0] = *(const us8*)p;
      a[mt].u[1] = *(const us8*)(p + 16);
    }
#pragma unroll
    for (int nt = 0; nt < 2; ++nt) {
      const unsigned short* p = Wbase + (size_t)nt * 16 * K + k0;
      b[nt].u[0] = *(const us8*)p;
      b[nt].u[1] = *(const us8*)(p + 16);
    }
    __builtin_prefetch(Abase + k0 + 64, 0, 1);   // global_prefetch_b8
    __builtin_prefetch(Wbase + k0 + 64, 0, 1);
#pragma unroll
    for (int mt = 0; mt < 4; ++mt)
#pragma unroll
      for (int nt = 0; nt < 2; ++nt)
        acc[mt][nt] = wmma_bf16(a[mt].v, b[nt].v, acc[mt][nt]);
  }

  // C/D layout: VGPR r holds row (r + 8*half), column = lane&15
#pragma unroll
  for (int mt = 0; mt < 4; ++mt) {
#pragma unroll
    for (int r = 0; r < 8; ++r) {
      const size_t row = (size_t)(bm + mt * 16 + r + 8 * hlf);
#pragma unroll
      for (int nt = 0; nt < 2; ++nt) {
        const size_t col = (size_t)(bn + nt * 16 + l16);
        const float  val = acc[mt][nt][r] * scale;
        if (OUT_F32) ((float*)C)[row * (size_t)N + col] = val;
        else         ((unsigned short*)C)[row * (size_t)N + col] = f2bf(val);
      }
    }
  }
}

// ---------------------------------------------------------------------------
// Fused GQA flash attention.
// Grid: B * 32 heads * (L/64) q-tiles. Block: 128 threads = 4 waves.
// Each wave owns a 16-row q strip; K/V tiles of 64 keys streamed into LDS via
// global_load_async_to_lds_b128 (ASYNCcnt), V transposed in-LDS afterwards.
// Q is pre-scaled by 1/sqrt(D) in the projection GEMM.
// ---------------------------------------------------------------------------
#define LSEQ 2048

__global__ __launch_bounds__(128)
void gqa_flash_attn(const unsigned short* __restrict__ Qb,   // [B*L, 2048]
                    const unsigned short* __restrict__ Kb,   // [B*L, 512]
                    const unsigned short* __restrict__ Vb,   // [B*L, 512]
                    unsigned short* __restrict__ Ob) {       // [B*L, 2048]
  __shared__ unsigned short sK[64 * 64];      // [keypos][d]
  __shared__ unsigned short sVrow[64 * 64];   // async staging: [keypos][d]
  __shared__ unsigned short sV[64 * 64];      // transposed: [d][keypos]
  __shared__ unsigned short sP[4][16 * 64];   // per-wave P scratch [row][keypos]

  const int lane = threadIdx.x & 31;
  const int wv   = threadIdx.x >> 5;
  const int hlf  = lane >> 4;
  const int l16  = lane & 15;

  const int bid = blockIdx.x;
  const int qt  = bid & 31;           // q tile (L/64 = 32)
  const int qh  = (bid >> 5) & 31;    // q head
  const int bb  = bid >> 10;          // batch
  const int kvh = qh >> 2;            // group of 4 q heads share one kv head

  const size_t qrow0 = (size_t)(bb * LSEQ + qt * 64 + wv * 16);

  // Q strip (16 x 64) as two A fragments (d 0..31 and 32..63), kept resident.
  ABFrag qf[2];
  {
    const unsigned short* p = Qb + (qrow0 + l16) * 2048 + qh * 64;
    qf[0].u[0] = *(const us8*)(p + hlf * 8);
    qf[0].u[1] = *(const us8*)(p + 16 + hlf * 8);
    qf[1].u[0] = *(const us8*)(p + 32 + hlf * 8);
    qf[1].u[1] = *(const us8*)(p + 48 + hlf * 8);
  }

  const v8f vzero = {0.f,0.f,0.f,0.f,0.f,0.f,0.f,0.f};
  v8f o[4];
#pragma unroll
  for (int t = 0; t < 4; ++t) o[t] = vzero;
  float mrun[8], lrun[8];
#pragma unroll
  for (int r = 0; r < 8; ++r) { mrun[r] = -3.0e38f; lrun[r] = 0.f; }

  const unsigned short* Kcol = Kb + (size_t)bb * LSEQ * 512 + kvh * 64;
  const unsigned short* Vcol = Vb + (size_t)bb * LSEQ * 512 + kvh * 64;

  for (int kt = 0; kt < LSEQ / 64; ++kt) {
    __syncthreads();   // LDS tiles from previous iteration fully consumed

    // Async-stage K and V tiles (row-major): 512 x 16B chunks, 128 threads.
#pragma unroll
    for (int it = 0; it < 4; ++it) {
      const int c  = it * 128 + threadIdx.x;  // 16-byte chunk id 0..511
      const int kp = c >> 3;
      const int d  = (c & 7) * 8;
      const size_t g = (size_t)(kt * 64 + kp) * 512 + d;
      copy16_g2lds(Kcol + g, &sK[kp * 64 + d]);
      copy16_g2lds(Vcol + g, &sVrow[kp * 64 + d]);
    }
    wait_async_copies();       // s_wait_asynccnt 0
    __syncthreads();

    // In-LDS transpose of V so PV B-fragments are contiguous over keypos.
#pragma unroll
    for (int it = 0; it < 8; ++it) {
      const int idx = (it * 128 + threadIdx.x) * 4;   // 0..4095
      const int kp  = idx >> 6;
      const int d   = idx & 63;
      us4 vvv = *(const us4*)&sVrow[kp * 64 + d];
      sV[(d + 0) * 64 + kp] = vvv[0];
      sV[(d + 1) * 64 + kp] = vvv[1];
      sV[(d + 2) * 64 + kp] = vvv[2];
      sV[(d + 3) * 64 + kp] = vvv[3];
    }
    __syncthreads();

    // S = Q K^T : 16 x 64 strip = 4 WMMA tiles, K-depth 64 = 2 chained WMMAs
    v8f s[4];
#pragma unroll
    for (int nt = 0; nt < 4; ++nt) {
      ABFrag k0f, k1f;
      const unsigned short* p = &sK[(nt * 16 + l16) * 64 + hlf * 8];
      k0f.u[0] = *(const us8*)p;
      k0f.u[1] = *(const us8*)(p + 16);
      k1f.u[0] = *(const us8*)(p + 32);
      k1f.u[1] = *(const us8*)(p + 48);
      v8f z = vzero;
      z = wmma_bf16(qf[0].v, k0f.v, z);
      z = wmma_bf16(qf[1].v, k1f.v, z);
      s[nt] = z;
    }

    // Online softmax in the C layout (row = r + 8*half, col across 16 lanes)
#pragma unroll
    for (int r = 0; r < 8; ++r) {
      float mx = fmaxf(fmaxf(s[0][r], s[1][r]), fmaxf(s[2][r], s[3][r]));
      mx = fmaxf(mx, __shfl_xor(mx, 1));
      mx = fmaxf(mx, __shfl_xor(mx, 2));
      mx = fmaxf(mx, __shfl_xor(mx, 4));
      mx = fmaxf(mx, __shfl_xor(mx, 8));
      const float mnew  = fmaxf(mrun[r], mx);
      const float alpha = __expf(mrun[r] - mnew);
      mrun[r] = mnew;
      float sum = 0.f;
#pragma unroll
      for (int t = 0; t < 4; ++t) {
        const float pv = __expf(s[t][r] - mnew);
        s[t][r] = pv;
        sum += pv;
      }
      sum += __shfl_xor(sum, 1);
      sum += __shfl_xor(sum, 2);
      sum += __shfl_xor(sum, 4);
      sum += __shfl_xor(sum, 8);
      lrun[r] = lrun[r] * alpha + sum;
#pragma unroll
      for (int t = 0; t < 4; ++t) o[t][r] *= alpha;
    }

    // Round-trip P through per-wave LDS to convert C layout -> A layout.
    unsigned short* pw = &sP[wv][0];
#pragma unroll
    for (int t = 0; t < 4; ++t)
#pragma unroll
      for (int r = 0; r < 8; ++r)
        pw[(r + 8 * hlf) * 64 + t * 16 + l16] = f2bf(s[t][r]);
    // same-wave DS ops are in-order; no block barrier needed (private region)

    ABFrag pf[2];
    const unsigned short* pr = &sP[wv][l16 * 64];
    pf[0].u[0] = *(const us8*)(pr + hlf * 8);
    pf[0].u[1] = *(const us8*)(pr + 16 + hlf * 8);
    pf[1].u[0] = *(const us8*)(pr + 32 + hlf * 8);
    pf[1].u[1] = *(const us8*)(pr + 48 + hlf * 8);

    // O += P V : 4 output d-tiles, K-depth 64 over key positions
#pragma unroll
    for (int nt = 0; nt < 4; ++nt) {
#pragma unroll
      for (int c = 0; c < 2; ++c) {
        ABFrag vf;
        const unsigned short* vp = &sV[(nt * 16 + l16) * 64 + c * 32 + hlf * 8];
        vf.u[0] = *(const us8*)vp;
        vf.u[1] = *(const us8*)(vp + 16);
        o[nt] = wmma_bf16(pf[c].v, vf.v, o[nt]);
      }
    }
  }

  // Normalize and emit bf16 for the O-projection GEMM.
#pragma unroll
  for (int r = 0; r < 8; ++r) {
    const float  rinv = 1.0f / lrun[r];
    const size_t row  = qrow0 + r + 8 * hlf;
#pragma unroll
    for (int t = 0; t < 4; ++t)
      Ob[row * 2048 + qh * 64 + t * 16 + l16] = f2bf(o[t][r] * rinv);
  }
}

// ---------------------------------------------------------------------------
// Host-side orchestration (graph-capture safe: only launches on `stream`)
// ---------------------------------------------------------------------------
extern "C" void kernel_launch(void* const* d_in, const int* in_sizes, int n_in,
                              void* d_out, int out_size, void* d_ws, size_t ws_size,
                              hipStream_t stream) {
  (void)in_sizes; (void)n_in; (void)out_size; (void)ws_size;
  const float* x  = (const float*)d_in[0];
  const float* WQ = (const float*)d_in[1];
  const float* WK = (const float*)d_in[2];
  const float* WV = (const float*)d_in[3];
  const float* WO = (const float*)d_in[4];

  const int B = 2, L = 2048, E = 2048, KV = 512;
  const size_t NX  = (size_t)B * L * E;   // x / q / attn-out elements
  const size_t NWQ = (size_t)E * E;
  const size_t NWK = (size_t)KV * E;
  const size_t NWV = (size_t)KV * E;
  const size_t NWO = (size_t)E * E;
  const size_t NKB = (size_t)B * L * KV;
  const size_t NVB = (size_t)B * L * KV;

  unsigned short* xb  = (unsigned short*)d_ws;
  unsigned short* wqb = xb  + NX;
  unsigned short* wkb = wqb + NWQ;
  unsigned short* wvb = wkb + NWK;
  unsigned short* wob = wvb + NWV;
  unsigned short* qb  = wob + NWO;
  unsigned short* kb  = qb  + NX;
  unsigned short* vb  = kb  + NKB;
  unsigned short* ab  = vb  + NVB;

  auto cvt = [&](const float* s, unsigned short* d, size_t n) {
    const int n4 = (int)(n / 4);
    cvt_f32_to_bf16<<<(n4 + 255) / 256, 256, 0, stream>>>(s, d, n4);
  };
  cvt(x,  xb,  NX);
  cvt(WQ, wqb, NWQ);
  cvt(WK, wkb, NWK);
  cvt(WV, wvb, NWV);
  cvt(WO, wob, NWO);

  const dim3 blk(256);
  // QKV projections; fold 1/sqrt(HEAD_DIM)=0.125 into Q.
  gemm_bf16_wmma<0><<<dim3(E  / 128, (B * L) / 128), blk, 0, stream>>>(
      xb, wqb, qb, B * L, E,  E, 0.125f);
  gemm_bf16_wmma<0><<<dim3(KV / 128, (B * L) / 128), blk, 0, stream>>>(
      xb, wkb, kb, B * L, KV, E, 1.0f);
  gemm_bf16_wmma<0><<<dim3(KV / 128, (B * L) / 128), blk, 0, stream>>>(
      xb, wvb, vb, B * L, KV, E, 1.0f);

  // Fused GQA attention: B * 32 heads * 32 q-tiles blocks of 4 waves.
  gqa_flash_attn<<<dim3(B * 32 * (L / 64)), dim3(128), 0, stream>>>(
      qb, kb, vb, ab);

  // Output projection -> f32 result.
  gemm_bf16_wmma<1><<<dim3(E / 128, (B * L) / 128), blk, 0, stream>>>(
      ab, wob, d_out, B * L, E, E, 1.0f);
}
